// LinearViolationAdaption_20650202759695
// MI455X (gfx1250) — matrix operand order repaired and checked by reading, not compiled, not run on hardware
//
#include <hip/hip_runtime.h>
#include <hip/hip_bf16.h>

typedef __attribute__((ext_vector_type(16))) _Float16 v16h;
typedef __attribute__((ext_vector_type(8)))  _Float16 v8h;
typedef __attribute__((ext_vector_type(4)))  _Float16 h4;
typedef __attribute__((ext_vector_type(2)))  _Float16 h2;
typedef __attribute__((ext_vector_type(8)))  float    v8f;
typedef __attribute__((ext_vector_type(4)))  float    f4;
typedef __attribute__((ext_vector_type(2)))  float    f2;

#define NDIM   512          // N (columns of A, length of x)
#define MDIM   512          // M (rows of A, length of b)
#define ALPHA  0.005f
#define SCALE  0.001f
#define DELTA  0.1f
#define ITERS  51           // MAX_ITER + 1

// ---------------------------------------------------------------------------
// Tiered one-time preprocessing:
//   cvt_f32_to_f16     : A -> Ah                 (134 MB, fits in 192 MB L2)
//   cvt_tr_f32_to_f16  : A -> Ah and AhT = A^T   (gradient becomes row-major)
// ---------------------------------------------------------------------------
__global__ __launch_bounds__(256) void cvt_f32_to_f16(
    const float* __restrict__ A, _Float16* __restrict__ Ah, long long n)
{
    long long i = ((long long)blockIdx.x * 256 + threadIdx.x) * 4;
    if (i + 3 < n) {
        f4 v = *(const f4*)(A + i);
        h4 o;
        o[0] = (_Float16)v[0]; o[1] = (_Float16)v[1];
        o[2] = (_Float16)v[2]; o[3] = (_Float16)v[3];
        *(h4*)(Ah + i) = o;
    }
}

// One 64x64 tile per workgroup; coalesced read, coalesced write of both Ah
// (same layout) and AhT (transposed, staged through padded LDS tile).
__global__ __launch_bounds__(256) void cvt_tr_f32_to_f16(
    const float* __restrict__ A, _Float16* __restrict__ Ah,
    _Float16* __restrict__ AhT)
{
    __shared__ _Float16 tile[64][65];            // +1 pad: no bank conflicts

    const int p    = blockIdx.x >> 6;            // problem
    const int tidx = blockIdx.x & 63;            // 8x8 tile grid
    const int tm   = (tidx >> 3) * 64;           // tile row origin (m)
    const int tn   = (tidx & 7) * 64;            // tile col origin (n)

    const float* Ap  = A   + (size_t)p * MDIM * NDIM;
    _Float16*    Hp  = Ah  + (size_t)p * MDIM * NDIM;
    _Float16*    HpT = AhT + (size_t)p * MDIM * NDIM;

    const int t  = threadIdx.x;
    const int rl = t >> 4;                       // 0..15
    const int c4 = (t & 15) * 4;                 // 0,4,..,60

    #pragma unroll
    for (int pass = 0; pass < 4; ++pass) {
        const int m = rl + pass * 16;
        const size_t idx = (size_t)(tm + m) * NDIM + tn + c4;
        f4 v = *(const f4*)(Ap + idx);
        h4 o;
        o[0] = (_Float16)v[0]; o[1] = (_Float16)v[1];
        o[2] = (_Float16)v[2]; o[3] = (_Float16)v[3];
        *(h4*)(Hp + idx) = o;
        tile[m][c4 + 0] = o[0]; tile[m][c4 + 1] = o[1];
        tile[m][c4 + 2] = o[2]; tile[m][c4 + 3] = o[3];
    }
    __syncthreads();

    const int nl = t >> 4;
    const int m4 = (t & 15) * 4;
    #pragma unroll
    for (int pass = 0; pass < 4; ++pass) {
        const int n = nl + pass * 16;
        h4 w;
        w[0] = tile[m4 + 0][n]; w[1] = tile[m4 + 1][n];
        w[2] = tile[m4 + 2][n]; w[3] = tile[m4 + 3][n];
        *(h4*)(HpT + (size_t)(tn + n) * MDIM + tm + m4) = w;
    }
}

// ---------------------------------------------------------------------------
// Broadcast-WMMA matvec: vout[512] = Amat(512x512 row-major f16) * vin[512].
// Each wave owns 4 blocks of 16 rows; x-chunk is replicated across the 16
// B-columns so every D column equals y; lanes 0/16 extract it.
// NT=true streams Amat with non-temporal hints (keeps Ah L2-resident while
// AhT streams from HBM).
// ---------------------------------------------------------------------------
template <bool NT>
__device__ __forceinline__ void wmma_matvec16(
    const _Float16* __restrict__ Amat, const _Float16* __restrict__ vin,
    float* __restrict__ vout, int wave, int lane, int g, int r)
{
    for (int blk = 0; blk < 4; ++blk) {
        const int rb = (wave * 4 + blk) * 16;
        const _Float16* Arow = Amat + (size_t)(rb + r) * NDIM;
        v8f acc = {};
        #pragma unroll 4
        for (int kc = 0; kc < NDIM; kc += 32) {
            // A-operand: lane L: elems 0-7 = K kc+g*16.., elems 8-15 = K kc+g*16+8..
            v8h lo, hi;
            if (NT) {
                lo = __builtin_nontemporal_load((const v8h*)(Arow + kc + g * 16));
                hi = __builtin_nontemporal_load((const v8h*)(Arow + kc + g * 16 + 8));
            } else {
                lo = *(const v8h*)(Arow + kc + g * 16);
                hi = *(const v8h*)(Arow + kc + g * 16 + 8);
            }
            v16h a;
            #pragma unroll
            for (int j = 0; j < 8; ++j) { a[j] = lo[j]; a[j + 8] = hi[j]; }
            // B-operand (vector broadcast over all 16 columns):
            // lane L, elem j = vin[kc + g*16 + j]
            v16h bv = *(const v16h*)(vin + kc + g * 16);
            acc = __builtin_amdgcn_wmma_f32_16x16x32_f16(
                      false, a, false, bv, (short)0, acc, false, false);
        }
        // D: lane 0 holds M=rb+0..7 (vgprs 0..7) at N=0; lane 16 holds M=rb+8..15
        if (lane == 0) {
            #pragma unroll
            for (int j = 0; j < 8; ++j) vout[rb + j] = acc[j];
        } else if (lane == 16) {
            #pragma unroll
            for (int j = 0; j < 8; ++j) vout[rb + 8 + j] = acc[j];
        }
    }
}

// ---------------------------------------------------------------------------
// Main solver: one workgroup (8 wave32s) per (b,s) problem.
// AhT != nullptr : gradient also via WMMA (dual-copy mode).
// AhT == nullptr : gradient via coalesced packed-f16 VALU.
// ---------------------------------------------------------------------------
__global__ __launch_bounds__(256) void solve_f16(
    const float* __restrict__ x0, const _Float16* __restrict__ Ah,
    const _Float16* __restrict__ AhT, const float* __restrict__ bIn,
    float* __restrict__ out)
{
    const int p = blockIdx.x;

    __shared__ __align__(32) float    xs[NDIM];
    __shared__ __align__(32) _Float16 xh[NDIM];
    __shared__ __align__(16) float    violf[MDIM];
    __shared__ __align__(32) _Float16 violh[MDIM];
    __shared__ __align__(16) float    gL[NDIM];
    __shared__ float bsh[MDIM];
    __shared__ float partial[256];
    __shared__ float totalS;

    const int t    = threadIdx.x;
    const int lane = t & 31;
    const int wave = t >> 5;
    const int g    = lane >> 4;   // half-wave group (0/1)
    const int r    = lane & 15;   // row within 16-row block

    const float*    xg  = x0  + (size_t)p * NDIM;
    const float*    bg  = bIn + (size_t)p * MDIM;
    const _Float16* Ap  = Ah  + (size_t)p * MDIM * NDIM;
    const _Float16* ApT = AhT ? AhT + (size_t)p * MDIM * NDIM : nullptr;

    for (int i = t; i < NDIM; i += 256) {
        float v = xg[i];
        xs[i] = v; xh[i] = (_Float16)v;
    }
    for (int i = t; i < MDIM; i += 256) bsh[i] = bg[i];
    __syncthreads();

    for (int iter = 0; iter < ITERS; ++iter) {
        // ---- Ax ------------------------------------------------------------
        wmma_matvec16<false>(Ap, xh, violf, wave, lane, g, r);
        __syncthreads();

        // ---- viol = max(Ax - b, 0); activity test --------------------------
        float ps = 0.f;
        for (int i = t; i < MDIM; i += 256) {
            float v = violf[i] - bsh[i];
            v = v > 0.f ? v : 0.f;
            violf[i] = v;
            violh[i] = (_Float16)v;
            ps += v;
        }
        partial[t] = ps;
        __syncthreads();
        if (t == 0) {                       // deterministic serial reduction
            float s = 0.f;
            for (int i = 0; i < 256; ++i) s += partial[i];
            totalS = s;
        }
        __syncthreads();
        if (totalS < DELTA) break;          // problem inactive: x frozen forever

        // ---- g = A^T viol --------------------------------------------------
        const int n0 = 2 * t;
        float g0, g1;
        if (ApT) {                          // uniform branch
            wmma_matvec16<true>(ApT, violh, gL, wave, lane, g, r);
            __syncthreads();
            g0 = gL[n0]; g1 = gL[n0 + 1];
        } else {
            g0 = 0.f; g1 = 0.f;
            const h2* colw = (const h2*)(Ap) + t;   // h2 idx m*256+t = halves m*512+2t
            #pragma unroll 8
            for (int m = 0; m < MDIM; ++m) {
                float vm = violf[m];
                h2 w = colw[(size_t)m * 256];
                g0 += vm * (float)w[0];
                g1 += vm * (float)w[1];
            }
        }

        // ---- adaptive-lr projected update ---------------------------------
        float lr0 = ALPHA / (1.f + SCALE * g0);
        float lr1 = ALPHA / (1.f + SCALE * g1);
        float nx0 = xs[n0]     - lr0 * g0; nx0 = nx0 > 0.f ? nx0 : 0.f;
        float nx1 = xs[n0 + 1] - lr1 * g1; nx1 = nx1 > 0.f ? nx1 : 0.f;
        xs[n0]     = nx0; xh[n0]     = (_Float16)nx0;
        xs[n0 + 1] = nx1; xh[n0 + 1] = (_Float16)nx1;
        __syncthreads();
    }

    for (int i = t; i < NDIM; i += 256) out[(size_t)p * NDIM + i] = xs[i];
}

// ---------------------------------------------------------------------------
// Fallback (workspace too small for any f16 copy): pure f32 VALU solver.
// ---------------------------------------------------------------------------
__global__ __launch_bounds__(256) void solve_f32(
    const float* __restrict__ x0, const float* __restrict__ A,
    const float* __restrict__ bIn, float* __restrict__ out)
{
    const int p = blockIdx.x;
    __shared__ float xs[NDIM];
    __shared__ float viol[MDIM];
    __shared__ float bsh[MDIM];
    __shared__ float partial[256];
    __shared__ float totalS;

    const int t = threadIdx.x;
    const float* Ap = A + (size_t)p * MDIM * NDIM;

    for (int i = t; i < NDIM; i += 256) xs[i]  = x0[(size_t)p * NDIM + i];
    for (int i = t; i < MDIM; i += 256) bsh[i] = bIn[(size_t)p * MDIM + i];
    __syncthreads();

    for (int iter = 0; iter < ITERS; ++iter) {
        for (int rr = 0; rr < 2; ++rr) {
            const int m = t + rr * 256;
            const float* Arow = Ap + (size_t)m * NDIM;
            float acc = 0.f;
            #pragma unroll 8
            for (int k = 0; k < NDIM; ++k) acc += Arow[k] * xs[k];
            viol[m] = acc;
        }
        __syncthreads();

        float ps = 0.f;
        for (int i = t; i < MDIM; i += 256) {
            float v = viol[i] - bsh[i];
            v = v > 0.f ? v : 0.f;
            viol[i] = v; ps += v;
        }
        partial[t] = ps;
        __syncthreads();
        if (t == 0) { float s = 0.f; for (int i = 0; i < 256; ++i) s += partial[i]; totalS = s; }
        __syncthreads();
        if (totalS < DELTA) break;

        const int n0 = 2 * t;
        float g0 = 0.f, g1 = 0.f;
        const f2* colw = (const f2*)(Ap) + t;
        #pragma unroll 8
        for (int m = 0; m < MDIM; ++m) {
            float vm = viol[m];
            f2 w = colw[(size_t)m * 256];
            g0 += vm * w[0];
            g1 += vm * w[1];
        }
        float lr0 = ALPHA / (1.f + SCALE * g0);
        float lr1 = ALPHA / (1.f + SCALE * g1);
        float nx0 = xs[n0]     - lr0 * g0; nx0 = nx0 > 0.f ? nx0 : 0.f;
        float nx1 = xs[n0 + 1] - lr1 * g1; nx1 = nx1 > 0.f ? nx1 : 0.f;
        xs[n0]     = nx0;
        xs[n0 + 1] = nx1;
        __syncthreads();
    }

    for (int i = t; i < NDIM; i += 256) out[(size_t)p * NDIM + i] = xs[i];
}

// ---------------------------------------------------------------------------
extern "C" void kernel_launch(void* const* d_in, const int* in_sizes, int n_in,
                              void* d_out, int out_size, void* d_ws, size_t ws_size,
                              hipStream_t stream)
{
    const float* x = (const float*)d_in[0];   // [B,S,N]
    const float* A = (const float*)d_in[1];   // [B,S,M,N]
    const float* b = (const float*)d_in[2];   // [B,S,M]
    float* out = (float*)d_out;

    const long long aelems = (long long)in_sizes[1];     // B*S*M*N
    const int nprob = in_sizes[2] / MDIM;                 // B*S problems
    const size_t fp16_bytes = (size_t)aelems * sizeof(_Float16);

    if (ws_size >= 2 * fp16_bytes) {
        // Dual-copy: Ah (L2-resident) + AhT (NT-streamed) -> all-WMMA solver.
        _Float16* Ah  = (_Float16*)d_ws;
        _Float16* AhT = Ah + aelems;
        cvt_tr_f32_to_f16<<<nprob * 64, 256, 0, stream>>>(A, Ah, AhT);
        solve_f16<<<nprob, 256, 0, stream>>>(x, Ah, AhT, b, out);
    } else if (ws_size >= fp16_bytes) {
        _Float16* Ah = (_Float16*)d_ws;
        const long long nvec = aelems / 4;
        const unsigned blocks = (unsigned)((nvec + 255) / 256);
        cvt_f32_to_f16<<<blocks, 256, 0, stream>>>(A, Ah, aelems);
        solve_f16<<<nprob, 256, 0, stream>>>(x, Ah, nullptr, b, out);
    } else {
        solve_f32<<<nprob, 256, 0, stream>>>(x, A, b, out);
    }
}
